// TransformerLayer_89670327206487
// MI455X (gfx1250) — matrix operand order repaired and checked by reading, not compiled, not run on hardware
//
#include <hip/hip_runtime.h>

// ---------------- problem constants ----------------
#define BB   4
#define SS   1024
#define DD   1024
#define HH   16
#define FFN  4096
#define DKH  64
#define BS_  (BB * SS)          // 4096 rows
#define EPSV 1e-6f
#define SCLQ 0.125f             // 1/sqrt(64)
#define KC   64                 // K-chunk staged in LDS per buffer

typedef __attribute__((ext_vector_type(16))) __bf16        bf16x16;
typedef __attribute__((ext_vector_type(8)))  float         f32x8;
typedef __attribute__((ext_vector_type(4)))  float         f32x4;
typedef __attribute__((ext_vector_type(4)))  unsigned int  u32x4;
typedef __attribute__((ext_vector_type(4)))  unsigned short u16x4;
typedef unsigned short u16_t;
typedef __attribute__((address_space(3))) u16_t lds_u16;

// ---------------- helpers ----------------
__device__ __forceinline__ u16_t f2bf(float f) {
  unsigned int u = __float_as_uint(f);
  u += 0x7FFFu + ((u >> 16) & 1u);          // round to nearest even
  return (u16_t)(u >> 16);
}

__device__ __forceinline__ bf16x16 ld_frag(const u16_t* p0, const u16_t* p1) {
  union { bf16x16 v; u32x4 q[2]; } u;
  u.q[0] = *reinterpret_cast<const u32x4*>(p0);
  u.q[1] = *reinterpret_cast<const u32x4*>(p1);
  return u.v;
}

__device__ __forceinline__ f32x8 wmma_bf16(bf16x16 a, bf16x16 b, f32x8 c) {
  return __builtin_amdgcn_wmma_f32_16x16x32_bf16(
      false, a, false, b, (short)0, c, false, false);
}

// async 16B copy global -> LDS (ASYNCcnt-tracked, CDNA5 GLOBAL_LOAD_ASYNC_TO_LDS_B128)
__device__ __forceinline__ void async_cp16(u16_t* dst_lds, const u16_t* src) {
  lds_u16* d3 = (lds_u16*)dst_lds;
  asm volatile("global_load_async_to_lds_b128 %0, %1, off"
               :: "v"(d3), "v"(src) : "memory");
}
__device__ __forceinline__ void wait_async0() {
  asm volatile("s_wait_asynccnt 0x0" ::: "memory");
}

__device__ __forceinline__ float gelu_f(float x) {
  const float k = 0.7978845608028654f;      // sqrt(2/pi)
  float t = tanhf(k * (x + 0.044715f * x * x * x));
  return 0.5f * x * (1.0f + t);
}

// ---------------- f32 -> bf16 conversion ----------------
__global__ __launch_bounds__(256)
void cvt_f32_to_bf16(const float* __restrict__ in, u16_t* __restrict__ out, int n4) {
  int i = blockIdx.x * 256 + threadIdx.x;
  if (i >= n4) return;
  f32x4 f = reinterpret_cast<const f32x4*>(in)[i];
  u16x4 o;
  o.x = f2bf(f.x); o.y = f2bf(f.y); o.z = f2bf(f.z); o.w = f2bf(f.w);
  reinterpret_cast<u16x4*>(out)[i] = o;
}

// ---------------- V transpose: [B*S, D] -> [B,H,DK,S] ----------------
__global__ __launch_bounds__(256)
void v_transpose(const u16_t* __restrict__ in, u16_t* __restrict__ out) {
  int idx = blockIdx.x * 256 + threadIdx.x;          // over B*S*D
  int d = idx & (DD - 1);
  int s = (idx >> 10) & (SS - 1);
  int b = idx >> 20;
  int h = d >> 6, dk = d & (DKH - 1);
  out[((size_t)((b * HH + h) * DKH + dk) << 10) + s] = in[idx];
}

// ---------------- WMMA GEMM:  C[M,N] = A[M,K] * W[N,K]^T + bias ----------------
// 256 threads / 8 waves, block tile 128x128, double-buffered async LDS staging.
__device__ __forceinline__
void store_tile(const f32x8& a, int m, int n, const float* bias,
                float* outF, u16_t* outB, int N, int hi, int lo) {
  const int col = n + lo;
  const float bv = bias ? bias[col] : 0.0f;
#pragma unroll
  for (int v = 0; v < 8; ++v) {
    const size_t idx = (size_t)(m + hi * 8 + v) * N + col;
    const float y = a[v] + bv;
    if (outF) outF[idx] = y;
    else      outB[idx] = f2bf(y);
  }
}

__global__ __launch_bounds__(256)
void gemm_bf16_nt(const u16_t* __restrict__ A, const u16_t* __restrict__ W,
                  const float* __restrict__ bias,
                  float* __restrict__ outF, u16_t* __restrict__ outB,
                  int M, int N, int K) {
  __shared__ __attribute__((aligned(16))) u16_t As[2][128 * KC];
  __shared__ __attribute__((aligned(16))) u16_t Bs[2][128 * KC];

  const int tid  = threadIdx.x;
  const int lane = tid & 31;
  const int wv   = tid >> 5;                 // 8 waves: 2 (M=64) x 4 (N=32)
  const int hi = lane >> 4, lo = lane & 15;
  const int wm = (wv & 1) * 64;
  const int wn = (wv >> 1) * 32;
  const int m0 = blockIdx.y * 128;
  const int n0 = blockIdx.x * 128;

  // staging: threads 0..127 copy A rows, 128..255 copy B rows (KC*2 = 128B/row)
  const int  srow = tid & 127;
  const bool isB  = tid >= 128;
  const u16_t* gsrc = isB ? (W + (size_t)(n0 + srow) * K)
                          : (A + (size_t)(m0 + srow) * K);
  u16_t* lrow = (isB ? Bs[0] : As[0]) + srow * KC;
  const size_t bufstride = (size_t)128 * KC;

  auto stage = [&](int buf, int k0) {
    const u16_t* s = gsrc + k0;
    u16_t* d = lrow + (size_t)buf * bufstride;
#pragma unroll
    for (int c = 0; c < KC / 8; ++c)
      async_cp16(d + c * 8, s + c * 8);
  };

  f32x8 acc[4][2];
#pragma unroll
  for (int i = 0; i < 4; ++i)
#pragma unroll
    for (int j = 0; j < 2; ++j) acc[i][j] = {};

  stage(0, 0);
  int buf = 0;
  for (int k0 = 0; k0 < K; k0 += KC) {
    wait_async0();            // own async copies landed
    __syncthreads();          // everyone's copies landed / prev reads done
    if (k0 + KC < K) stage(buf ^ 1, k0 + KC);

    const u16_t* Ab = As[buf];
    const u16_t* Bb = Bs[buf];
#pragma unroll
    for (int kk = 0; kk < KC; kk += 32) {
      bf16x16 af[4], bfr[2];
#pragma unroll
      for (int mt = 0; mt < 4; ++mt) {
        const u16_t* p = Ab + (wm + mt * 16 + lo) * KC + kk + (hi << 3);
        af[mt] = ld_frag(p, p + 16);
      }
#pragma unroll
      for (int nt = 0; nt < 2; ++nt) {
        const u16_t* p = Bb + (wn + nt * 16 + lo) * KC + kk + (hi << 3);
        bfr[nt] = ld_frag(p, p + 16);
      }
#pragma unroll
      for (int mt = 0; mt < 4; ++mt)
#pragma unroll
        for (int nt = 0; nt < 2; ++nt)
          acc[mt][nt] = wmma_bf16(af[mt], bfr[nt], acc[mt][nt]);
    }
    buf ^= 1;
  }

#pragma unroll
  for (int mt = 0; mt < 4; ++mt)
#pragma unroll
    for (int nt = 0; nt < 2; ++nt)
      store_tile(acc[mt][nt], m0 + wm + mt * 16, n0 + wn + nt * 16,
                 bias, outF, outB, N, hi, lo);
}

// ---------------- flash attention (per (b,h), 64 query rows / block) ----------------
__global__ __launch_bounds__(128)
void flash_attn(const u16_t* __restrict__ Qb, const u16_t* __restrict__ Kb,
                const u16_t* __restrict__ Vt, const int* __restrict__ mask,
                u16_t* __restrict__ Ob) {
  __shared__ __attribute__((aligned(16))) float  sc[4][16][32];
  __shared__ __attribute__((aligned(16))) u16_t  pbuf[4][16][32];
  __shared__ float mrow[4][16], lrow[4][16], arow[4][16];

  const int lane = threadIdx.x & 31;
  const int w = threadIdx.x >> 5;          // wave id: 16 query rows each
  const int hi = lane >> 4, lo = lane & 15;
  const int h = blockIdx.y;
  const int b = blockIdx.z;
  const int q0 = blockIdx.x * 64 + w * 16;

  const u16_t* qp = Qb + (size_t)(b * SS + q0 + lo) * DD + h * DKH + (hi << 3);
  bf16x16 qa0 = ld_frag(qp,      qp + 16);
  bf16x16 qa1 = ld_frag(qp + 32, qp + 48);

  f32x8 o0 = {}, o1 = {}, o2 = {}, o3 = {};
  if (lane < 16) { mrow[w][lane] = -1e30f; lrow[w][lane] = 0.0f; }
  __syncthreads();

  for (int kb0 = 0; kb0 < SS; kb0 += 32) {
    // ---- scores S = Q*K^T over two 16-key tiles (contraction = head dim 64)
    const u16_t* kp0 = Kb + (size_t)(b * SS + kb0 + lo) * DD + h * DKH + (hi << 3);
    const u16_t* kp1 = kp0 + (size_t)16 * DD;
    f32x8 s0 = {}, s1 = {};
    s0 = wmma_bf16(qa0, ld_frag(kp0,      kp0 + 16), s0);
    s0 = wmma_bf16(qa1, ld_frag(kp0 + 32, kp0 + 48), s0);
    s1 = wmma_bf16(qa0, ld_frag(kp1,      kp1 + 16), s1);
    s1 = wmma_bf16(qa1, ld_frag(kp1 + 32, kp1 + 48), s1);

    const int mk0 = mask[b * SS + kb0 + lo];
    const int mk1 = mask[b * SS + kb0 + 16 + lo];
#pragma unroll
    for (int v = 0; v < 8; ++v) {
      const int r = hi * 8 + v;                    // C-layout row
      sc[w][r][lo]      = mk0 ? s0[v] * SCLQ : -1e9f;
      sc[w][r][lo + 16] = mk1 ? s1[v] * SCLQ : -1e9f;
    }
    __syncthreads();

    // ---- online softmax over this 32-key block (one row per lane 0..15)
    if (lane < 16) {
      float mold = mrow[w][lane], lold = lrow[w][lane];
      float mx = mold;
#pragma unroll
      for (int j = 0; j < 32; ++j) mx = fmaxf(mx, sc[w][lane][j]);
      float alpha = __expf(mold - mx);
      float sum = 0.0f;
#pragma unroll
      for (int j = 0; j < 32; ++j) {
        float pv = __expf(sc[w][lane][j] - mx);
        pbuf[w][lane][j] = f2bf(pv);
        sum += pv;
      }
      mrow[w][lane] = mx;
      lrow[w][lane] = lold * alpha + sum;
      arow[w][lane] = alpha;
    }
    __syncthreads();

    // ---- rescale O accumulators by alpha(row)
    float al[8];
#pragma unroll
    for (int v = 0; v < 8; ++v) al[v] = arow[w][hi * 8 + v];
#pragma unroll
    for (int v = 0; v < 8; ++v) {
      o0[v] *= al[v]; o1[v] *= al[v]; o2[v] *= al[v]; o3[v] *= al[v];
    }

    // ---- O += P * V  (P A-frag from LDS, V B-frags contiguous from Vt)
    const u16_t* pp = &pbuf[w][lo][hi << 3];
    bf16x16 pf = ld_frag(pp, pp + 16);
    const u16_t* vb = Vt + (size_t)((b * HH + h) * DKH + lo) * SS + kb0 + (hi << 3);
    o0 = wmma_bf16(pf, ld_frag(vb,                   vb + 16),                   o0);
    o1 = wmma_bf16(pf, ld_frag(vb + (size_t)16 * SS, vb + (size_t)16 * SS + 16), o1);
    o2 = wmma_bf16(pf, ld_frag(vb + (size_t)32 * SS, vb + (size_t)32 * SS + 16), o2);
    o3 = wmma_bf16(pf, ld_frag(vb + (size_t)48 * SS, vb + (size_t)48 * SS + 16), o3);
  }

  // ---- epilogue: scale by 1/l, write bf16 to [B*S, D] head slice
  float il[8];
#pragma unroll
  for (int v = 0; v < 8; ++v) {
    float lv = lrow[w][hi * 8 + v];
    il[v] = (lv > 0.0f) ? 1.0f / lv : 0.0f;
  }
  u16_t* ob = Ob + (size_t)(b * SS + q0) * DD + h * DKH;
#pragma unroll
  for (int v = 0; v < 8; ++v) {
    const size_t rb = (size_t)(hi * 8 + v) * DD;
    ob[rb + lo]      = f2bf(o0[v] * il[v]);
    ob[rb + 16 + lo] = f2bf(o1[v] * il[v]);
    ob[rb + 32 + lo] = f2bf(o2[v] * il[v]);
    ob[rb + 48 + lo] = f2bf(o3[v] * il[v]);
  }
}

// ---------------- residual + LayerNorm (optional GELU on x) ----------------
__global__ __launch_bounds__(256)
void ln_residual(const float* __restrict__ x, const float* __restrict__ res,
                 const float* __restrict__ gam, const float* __restrict__ bet,
                 float* __restrict__ outF, u16_t* __restrict__ outB, int applyGelu) {
  __shared__ float red[256];
  const int tid = threadIdx.x;
  const size_t row = blockIdx.x;
  const float* xr = x + row * DD;
  const float* rr = res + row * DD;

  float vals[4];
  float s = 0.0f;
#pragma unroll
  for (int i = 0; i < 4; ++i) {
    int c = tid + i * 256;
    float v = xr[c];
    if (applyGelu) v = gelu_f(v);
    v += rr[c];
    vals[i] = v;
    s += v;
  }
  red[tid] = s; __syncthreads();
  for (int off = 128; off > 0; off >>= 1) {
    if (tid < off) red[tid] += red[tid + off];
    __syncthreads();
  }
  const float mean = red[0] / (float)DD;
  __syncthreads();

  float s2 = 0.0f;
#pragma unroll
  for (int i = 0; i < 4; ++i) { float d = vals[i] - mean; s2 += d * d; }
  red[tid] = s2; __syncthreads();
  for (int off = 128; off > 0; off >>= 1) {
    if (tid < off) red[tid] += red[tid + off];
    __syncthreads();
  }
  const float var = red[0] / (float)(DD - 1);        // unbiased, like torch
  const float inv = 1.0f / (sqrtf(var) + EPSV);

#pragma unroll
  for (int i = 0; i < 4; ++i) {
    int c = tid + i * 256;
    float y = gam[c] * (vals[i] - mean) * inv + bet[c];
    if (outF) outF[row * DD + c] = y;
    if (outB) outB[row * DD + c] = f2bf(y);
  }
}

// ---------------- host driver ----------------
extern "C" void kernel_launch(void* const* d_in, const int* in_sizes, int n_in,
                              void* d_out, int out_size, void* d_ws, size_t ws_size,
                              hipStream_t stream) {
  (void)in_sizes; (void)n_in; (void)out_size; (void)ws_size;

  const float* query = (const float*)d_in[0];
  // d_in[1] (key), d_in[2] (value) unused: reference is self-attention on query
  const int*   mask  = (const int*)d_in[3];
  const float* Wq = (const float*)d_in[4],  *bq = (const float*)d_in[5];
  const float* Wk = (const float*)d_in[6],  *bk = (const float*)d_in[7];
  const float* Wv = (const float*)d_in[8],  *bv = (const float*)d_in[9];
  const float* Wo = (const float*)d_in[10], *bo = (const float*)d_in[11];
  const float* W1 = (const float*)d_in[12], *b1 = (const float*)d_in[13];
  const float* W2 = (const float*)d_in[14], *b2 = (const float*)d_in[15];
  const float* gm1 = (const float*)d_in[16], *be1 = (const float*)d_in[17];
  const float* gm2 = (const float*)d_in[18], *be2 = (const float*)d_in[19];

  char* p = (char*)d_ws;
  auto carve = [&](size_t bytes) -> void* {
    void* r = (void*)p;
    p += (bytes + 255) & ~(size_t)255;
    return r;
  };
  u16_t* Xb  = (u16_t*)carve((size_t)BS_ * DD * 2);
  u16_t* Wqb = (u16_t*)carve((size_t)DD * DD * 2);
  u16_t* Wkb = (u16_t*)carve((size_t)DD * DD * 2);
  u16_t* Wvb = (u16_t*)carve((size_t)DD * DD * 2);
  u16_t* Wob = (u16_t*)carve((size_t)DD * DD * 2);
  u16_t* W1b = (u16_t*)carve((size_t)FFN * DD * 2);
  u16_t* W2b = (u16_t*)carve((size_t)DD * FFN * 2);
  u16_t* Qb  = (u16_t*)carve((size_t)BS_ * DD * 2);
  u16_t* Kb  = (u16_t*)carve((size_t)BS_ * DD * 2);
  u16_t* Vb  = (u16_t*)carve((size_t)BS_ * DD * 2);
  u16_t* Vt  = (u16_t*)carve((size_t)BS_ * DD * 2);
  u16_t* Ab  = (u16_t*)carve((size_t)BS_ * DD * 2);
  float* X1  = (float*)carve((size_t)BS_ * DD * 4);
  float* L1f = (float*)carve((size_t)BS_ * DD * 4);
  u16_t* L1b = (u16_t*)carve((size_t)BS_ * DD * 2);
  u16_t* H1b = (u16_t*)carve((size_t)BS_ * FFN * 2);
  float* H2  = (float*)carve((size_t)BS_ * DD * 4);

  // 1) convert activations + weights to bf16
  cvt_f32_to_bf16<<<(BS_ * DD) / 1024, 256, 0, stream>>>(query, Xb, (BS_ * DD) / 4);
  cvt_f32_to_bf16<<<(DD * DD) / 1024, 256, 0, stream>>>(Wq, Wqb, (DD * DD) / 4);
  cvt_f32_to_bf16<<<(DD * DD) / 1024, 256, 0, stream>>>(Wk, Wkb, (DD * DD) / 4);
  cvt_f32_to_bf16<<<(DD * DD) / 1024, 256, 0, stream>>>(Wv, Wvb, (DD * DD) / 4);
  cvt_f32_to_bf16<<<(DD * DD) / 1024, 256, 0, stream>>>(Wo, Wob, (DD * DD) / 4);
  cvt_f32_to_bf16<<<(FFN * DD) / 1024, 256, 0, stream>>>(W1, W1b, (FFN * DD) / 4);
  cvt_f32_to_bf16<<<(DD * FFN) / 1024, 256, 0, stream>>>(W2, W2b, (DD * FFN) / 4);

  // 2) QKV projections (bf16 outputs)
  dim3 gqkv(DD / 128, BS_ / 128);
  gemm_bf16_nt<<<gqkv, 256, 0, stream>>>(Xb, Wqb, bq, nullptr, Qb, BS_, DD, DD);
  gemm_bf16_nt<<<gqkv, 256, 0, stream>>>(Xb, Wkb, bk, nullptr, Kb, BS_, DD, DD);
  gemm_bf16_nt<<<gqkv, 256, 0, stream>>>(Xb, Wvb, bv, nullptr, Vb, BS_, DD, DD);

  // 3) transpose V for contiguous P*V B-fragments
  v_transpose<<<(BS_ * DD) / 256, 256, 0, stream>>>(Vb, Vt);

  // 4) fused flash attention
  dim3 gat(SS / 64, HH, BB);
  flash_attn<<<gat, 128, 0, stream>>>(Qb, Kb, Vt, mask, Ab);

  // 5) output projection, residual + LN1
  gemm_bf16_nt<<<gqkv, 256, 0, stream>>>(Ab, Wob, bo, X1, nullptr, BS_, DD, DD);
  ln_residual<<<BS_, 256, 0, stream>>>(X1, query, gm1, be1, L1f, L1b, 0);

  // 6) FFN: fc1 (bf16 out), fc2 (f32 out), then gelu + residual + LN2
  dim3 gf1(FFN / 128, BS_ / 128);
  gemm_bf16_nt<<<gf1, 256, 0, stream>>>(L1b, W1b, b1, nullptr, H1b, BS_, FFN, DD);
  dim3 gf2(DD / 128, BS_ / 128);
  gemm_bf16_nt<<<gf2, 256, 0, stream>>>(H1b, W2b, b2, H2, nullptr, BS_, DD, FFN);
  ln_residual<<<BS_, 256, 0, stream>>>(H2, L1f, gm2, be2, (float*)d_out, nullptr, 1);
}
